// Decoder_62010737819886
// MI455X (gfx1250) — compile-verified, matrix operand order checked by reading, not run on hardware
//
#include <hip/hip_runtime.h>
#include <hip/hip_bf16.h>

typedef __attribute__((ext_vector_type(16))) __bf16 v16bf;
typedef __attribute__((ext_vector_type(8)))  __bf16 v8bf;
typedef __attribute__((ext_vector_type(8)))  float  v8f;

#define BB 64
#define TT 512
#define EE 1024
#define HH 1024
#define AA 2048
#define KX 2048   // H + E
#define NBLK 128
#define NTHR 256

// ---- workspace layout (bytes) ----
#define OFF_W1T   0UL                          // [2048][2048] bf16 (W1 transposed)
#define OFF_W2T   (OFF_W1T  + 2048UL*2048*2)   // [512][2048]  bf16 (W2 transposed)
#define OFF_W34T  (OFF_W2T  + 512UL*2048*2)    // [1024][2048] bf16 ([Wih;Whh] transposed)
#define OFF_U     (OFF_W34T + 1024UL*2048*2)   // [64][2048] bf16  u = tanh(x@W1+b1)
#define OFF_S     (OFF_U    + 64UL*2048*2)     // [2][64][1024] bf16 ping-pong state
#define OFF_HB    (OFF_S    + 2UL*64*1024*2)   // [64][1024] bf16  h_t converted
#define OFF_CB    (OFF_HB   + 64UL*1024*2)     // [64][1024] bf16  context c
#define OFF_A     (OFF_CB   + 64UL*1024*2)     // [64][512] f32    alignment scores
#define OFF_BAR   (OFF_A    + 64UL*512*4)      // 2 x u32 barrier

// ---------------- WMMA helpers ----------------
// 16-bit A (16x32) and B (32x16) fragments share the same per-lane pattern:
// row = lane&15; lanes<16 hold K = {kb+0..7, kb+16..23}; lanes>=16 hold
// K = {kb+8..15, kb+24..31}  ->  two contiguous 16-byte loads per lane.
__device__ __forceinline__ v16bf load_frag(const __bf16* tile, int ld, int kbase, int lane) {
  const int r   = lane & 15;
  const int klo = (lane >> 4) << 3;
  const __bf16* p = tile + (size_t)r * ld + kbase + klo;
  v8bf lo = *(const v8bf*)(p);
  v8bf hi = *(const v8bf*)(p + 16);
  v16bf f;
#pragma unroll
  for (int i = 0; i < 8; ++i) { f[i] = lo[i]; f[8 + i] = hi[i]; }
  return f;
}

__device__ __forceinline__ v8f wmma_bf16(v16bf a, v16bf b, v8f c) {
  return __builtin_amdgcn_wmma_f32_16x16x32_bf16(false, a, false, b, (short)0, c, false, false);
}

// ---------------- grid-wide barrier ----------------
__device__ __forceinline__ void gsync(unsigned* bar) {
  __syncthreads();
  if (threadIdx.x == 0) {
    __threadfence();
    volatile unsigned* cnt = bar;
    volatile unsigned* gen = bar + 1;
    const unsigned g = *gen;
    if (atomicAdd((unsigned*)bar, 1u) == (unsigned)(NBLK - 1)) {
      *cnt = 0u;
      __threadfence();
      atomicAdd((unsigned*)(bar + 1), 1u);
    } else {
      while (*gen == g) { __builtin_amdgcn_s_sleep(2); }
    }
    __threadfence();
  }
  __syncthreads();
}

// ---------------- pack kernels (run once per call) ----------------
__global__ void pack_w1(const float* __restrict__ W1, __bf16* __restrict__ W1T) {
  const int i = blockIdx.x * blockDim.x + threadIdx.x;     // i = n*2048 + k
  const int n = i >> 11, k = i & 2047;
  W1T[i] = (__bf16)W1[(size_t)k * AA + n];
}
__global__ void pack_w2(const float* __restrict__ W2, __bf16* __restrict__ W2T) {
  const int i = blockIdx.x * blockDim.x + threadIdx.x;     // i = t*2048 + k
  const int t = i >> 11, k = i & 2047;
  W2T[i] = (__bf16)W2[(size_t)k * TT + t];
}
__global__ void pack_w34(const float* __restrict__ Wih, const float* __restrict__ Whh,
                         __bf16* __restrict__ W34T) {
  const int i = blockIdx.x * blockDim.x + threadIdx.x;     // i = n*2048 + k
  const int n = i >> 11, k = i & 2047;
  W34T[i] = (k < HH) ? (__bf16)Wih[(size_t)k * HH + n]
                     : (__bf16)Whh[(size_t)(k - HH) * HH + n];
}
__global__ void init_state(const float* __restrict__ h, __bf16* __restrict__ sbuf,
                           __bf16* __restrict__ hbb, unsigned* __restrict__ bar) {
  const int i = blockIdx.x * blockDim.x + threadIdx.x;     // 64*1024 elems
  const int b = i >> 10, col = i & 1023;
  sbuf[i] = (__bf16)0.0f;
  hbb[i]  = (__bf16)h[((size_t)b * TT + 0) * EE + col];
  if (i < 2) bar[i] = 0u;
}

// ---------------- persistent decoder ----------------
__global__ void __launch_bounds__(NTHR, 1)
decoder_persistent(const float* __restrict__ h,
                   const float* __restrict__ b1, const float* __restrict__ b2,
                   const float* __restrict__ bih, const float* __restrict__ bhh,
                   const __bf16* __restrict__ W1T, const __bf16* __restrict__ W2T,
                   const __bf16* __restrict__ W34T,
                   __bf16* __restrict__ u, __bf16* __restrict__ sbuf,
                   __bf16* __restrict__ hbb, __bf16* __restrict__ cbb,
                   float* __restrict__ abuf, unsigned* __restrict__ bar,
                   float* __restrict__ out) {
  __shared__ float lds[1792];
  const int tid  = threadIdx.x;
  const int lane = tid & 31;
  const int wave = tid >> 5;
  const int wg   = blockIdx.x;

#pragma clang loop unroll(disable)
  for (int t = 0; t < TT; ++t) {
    __bf16* scur = sbuf + (size_t)(t & 1) * (64 * 1024);
    __bf16* snxt = sbuf + (size_t)((t + 1) & 1) * (64 * 1024);

    // ---- phase 1: u = tanh([s|h_t] @ W1 + b1)  [64,2048] ----
    {
      const int n0 = wg * 16;
      const int mt = wave & 3, kh = wave >> 2;
      const int m0 = mt * 16;
      const __bf16* At = (kh == 0) ? (scur + (size_t)m0 * HH) : (hbb + (size_t)m0 * EE);
      const __bf16* Bt = W1T + (size_t)n0 * KX;
      const int kg0 = kh * 1024;
      v8f acc = {};
#pragma unroll 4
      for (int kb = 0; kb < 1024; kb += 32)
        acc = wmma_bf16(load_frag(At, 1024, kb, lane), load_frag(Bt, KX, kg0 + kb, lane), acc);
      if (kh == 1) {
#pragma unroll
        for (int r = 0; r < 8; ++r) lds[mt * 256 + lane * 8 + r] = acc[r];
      }
      __syncthreads();
      if (kh == 0) {
#pragma unroll
        for (int r = 0; r < 8; ++r) acc[r] += lds[mt * 256 + lane * 8 + r];
        const int n = n0 + (lane & 15);
        const float bias = b1[n];
        const int mrow = m0 + ((lane >> 4) << 3);
#pragma unroll
        for (int r = 0; r < 8; ++r)
          u[(size_t)(mrow + r) * AA + n] = (__bf16)tanhf(acc[r] + bias);
      }
    }
    gsync(bar);

    // ---- phase 2: a = u @ W2 + b2  [64,512] ----
    {
      const int m0 = (wg >> 5) * 16, n0 = (wg & 31) * 16;
      const __bf16* At = u + (size_t)m0 * AA;
      const __bf16* Bt = W2T + (size_t)n0 * AA;
      v8f acc = {};
      const int k0 = wave * 256;
#pragma unroll 4
      for (int kb = 0; kb < 256; kb += 32)
        acc = wmma_bf16(load_frag(At, AA, k0 + kb, lane), load_frag(Bt, AA, k0 + kb, lane), acc);
      if (wave > 0) {
#pragma unroll
        for (int r = 0; r < 8; ++r) lds[(wave - 1) * 256 + lane * 8 + r] = acc[r];
      }
      __syncthreads();
      if (wave == 0) {
        for (int w = 1; w < 8; ++w)
#pragma unroll
          for (int r = 0; r < 8; ++r) acc[r] += lds[(w - 1) * 256 + lane * 8 + r];
        const int n = n0 + (lane & 15);
        const float bias = b2[n];
        const int mrow = m0 + ((lane >> 4) << 3);
#pragma unroll
        for (int r = 0; r < 8; ++r) abuf[(size_t)(mrow + r) * TT + n] = acc[r] + bias;
      }
    }
    gsync(bar);

    // ---- phase 3: softmax(a) + context  c = e @ h_b ----
    {
      const int b = wg >> 1, half = wg & 1;
      float* sm  = lds;        // 512
      float* red = lds + 512;  // 256
      sm[tid]       = abuf[b * TT + tid];
      sm[tid + 256] = abuf[b * TT + tid + 256];
      __syncthreads();
      red[tid] = fmaxf(sm[tid], sm[tid + 256]);
      __syncthreads();
      for (int s = 128; s > 0; s >>= 1) {
        if (tid < s) red[tid] = fmaxf(red[tid], red[tid + s]);
        __syncthreads();
      }
      const float mx = red[0];
      __syncthreads();
      const float e0 = __expf(sm[tid] - mx), e1 = __expf(sm[tid + 256] - mx);
      red[tid] = e0 + e1;
      __syncthreads();
      for (int s = 128; s > 0; s >>= 1) {
        if (tid < s) red[tid] += red[tid + s];
        __syncthreads();
      }
      const float inv = 1.0f / red[0];
      __syncthreads();
      sm[tid] = e0 * inv;
      sm[tid + 256] = e1 * inv;
      __syncthreads();
      const int col = half * 512 + tid * 2;
      const float* hb = h + (size_t)b * TT * EE + col;
      float a0 = 0.0f, a1 = 0.0f;
#pragma unroll 4
      for (int tt = 0; tt < TT; ++tt) {
        const float w = sm[tt];
        const float2 v = *(const float2*)(hb + (size_t)tt * EE);
        a0 += w * v.x;
        a1 += w * v.y;
      }
      cbb[(size_t)b * EE + col]     = (__bf16)a0;
      cbb[(size_t)b * EE + col + 1] = (__bf16)a1;
    }
    gsync(bar);

    // ---- phase 4: s' = tanh([c|s] @ [Wih;Whh] + bih + bhh); also stage h_{t+1} ----
    {
      const int mhalf = wg & 1;
      const int n0 = (wg >> 1) * 16;
      const int mtl = wave & 1, kq = wave >> 1;
      const int m0 = (mhalf * 2 + mtl) * 16;
      const __bf16* At = (kq < 2) ? (cbb + (size_t)m0 * EE) : (scur + (size_t)m0 * HH);
      const __bf16* Bt = W34T + (size_t)n0 * KX;
      const int kloc0 = (kq < 2) ? kq * 512 : (kq - 2) * 512;
      const int kg0 = kq * 512;
      v8f acc = {};
#pragma unroll 4
      for (int kk = 0; kk < 512; kk += 32)
        acc = wmma_bf16(load_frag(At, 1024, kloc0 + kk, lane),
                        load_frag(Bt, KX, kg0 + kk, lane), acc);
      if (kq > 0) {
#pragma unroll
        for (int r = 0; r < 8; ++r) lds[((kq - 1) * 2 + mtl) * 256 + lane * 8 + r] = acc[r];
      }
      __syncthreads();
      if (kq == 0) {
        for (int q = 1; q < 4; ++q)
#pragma unroll
          for (int r = 0; r < 8; ++r) acc[r] += lds[((q - 1) * 2 + mtl) * 256 + lane * 8 + r];
        const int n = n0 + (lane & 15);
        const float bias = bih[n] + bhh[n];
        const int mrow = m0 + ((lane >> 4) << 3);
#pragma unroll
        for (int r = 0; r < 8; ++r) {
          const float sv = tanhf(acc[r] + bias);
          const int m = mrow + r;
          out[((size_t)m * TT + t) * HH + n] = sv;
          snxt[(size_t)m * HH + n] = (__bf16)sv;
        }
      }
      // stage h_{t+1} as bf16 (2 elems / thread across the grid)
      if (t + 1 < TT) {
        const int idx = (wg * NTHR + tid) * 2;   // 0 .. 65534
        const int bb = idx >> 10, col = idx & 1023;
        const float2 hv = *(const float2*)(h + ((size_t)bb * TT + (t + 1)) * EE + col);
        hbb[(size_t)bb * EE + col]     = (__bf16)hv.x;
        hbb[(size_t)bb * EE + col + 1] = (__bf16)hv.y;
      }
    }
    gsync(bar);
  }
}

extern "C" void kernel_launch(void* const* d_in, const int* in_sizes, int n_in,
                              void* d_out, int out_size, void* d_ws, size_t ws_size,
                              hipStream_t stream) {
  (void)in_sizes; (void)n_in; (void)out_size; (void)ws_size;
  const float* h   = (const float*)d_in[0];
  const float* W1  = (const float*)d_in[1];
  const float* b1  = (const float*)d_in[2];
  const float* W2  = (const float*)d_in[3];
  const float* b2  = (const float*)d_in[4];
  const float* Wih = (const float*)d_in[5];
  const float* Whh = (const float*)d_in[6];
  const float* bih = (const float*)d_in[7];
  const float* bhh = (const float*)d_in[8];

  char* ws = (char*)d_ws;
  __bf16*  W1T  = (__bf16*)(ws + OFF_W1T);
  __bf16*  W2T  = (__bf16*)(ws + OFF_W2T);
  __bf16*  W34T = (__bf16*)(ws + OFF_W34T);
  __bf16*  u    = (__bf16*)(ws + OFF_U);
  __bf16*  sbuf = (__bf16*)(ws + OFF_S);
  __bf16*  hbb  = (__bf16*)(ws + OFF_HB);
  __bf16*  cbb  = (__bf16*)(ws + OFF_CB);
  float*   abuf = (float*) (ws + OFF_A);
  unsigned* bar = (unsigned*)(ws + OFF_BAR);

  pack_w1 <<<16384, 256, 0, stream>>>(W1, W1T);
  pack_w2 <<< 4096, 256, 0, stream>>>(W2, W2T);
  pack_w34<<< 8192, 256, 0, stream>>>(Wih, Whh, W34T);
  init_state<<<256, 256, 0, stream>>>(h, sbuf, hbb, bar);

  decoder_persistent<<<NBLK, NTHR, 0, stream>>>(h, b1, b2, bih, bhh,
                                                W1T, W2T, W34T,
                                                u, sbuf, hbb, cbb, abuf, bar,
                                                (float*)d_out);
}
